// DeepseekV2LiteMoEWithGroupGeMM_13675175870989
// MI455X (gfx1250) — matrix operand order repaired and checked by reading, not compiled, not run on hardware
//
#include <hip/hip_runtime.h>
#include <hip/hip_bf16.h>

// ---------------- problem constants ----------------
#define H_DIM   2048
#define F_DIM   1408
#define NE      64
#define TOPK    6
#define CAP_E   512
#define FSH     2816
#define NTOK    2048

typedef __bf16 bf16;
typedef __attribute__((ext_vector_type(16))) __bf16 v16bf;
typedef __attribute__((ext_vector_type(8)))  float  v8f;
typedef int v4i __attribute__((vector_size(16)));   // matches async builtin's pointee

#define LDT 40   // LDS row stride (elements) for 32-wide K tiles; 16B aligned rows, bank-skewed

#define AS1 __attribute__((address_space(1)))
#define AS3 __attribute__((address_space(3)))

#if defined(__has_builtin)
#if __has_builtin(__builtin_amdgcn_global_load_async_to_lds_b128) && \
    __has_builtin(__builtin_amdgcn_s_wait_asynccnt)
#define USE_ASYNC 1
#endif
#endif
#ifndef USE_ASYNC
#define USE_ASYNC 0
#endif

// ---------------- WMMA fragment loads from LDS ----------------
// A (16x32 bf16): lanes 0-15 row M=lane, K=0..7 & 16..23 ; lanes 16-31: K=8..15 & 24..31
__device__ __forceinline__ v16bf frag_a(const bf16* As, int lane) {
    const bf16* p = As + (lane & 15) * LDT + ((lane >> 4) << 3);
    v16bf a;
    ((uint4*)&a)[0] = *(const uint4*)p;
    ((uint4*)&a)[1] = *(const uint4*)(p + 16);
    return a;
}
// B (32x16 bf16): lanes 0-15 col N=lane, K=0..7 & 16..23 ; lanes 16-31 col N=lane-16, K=8..15 & 24..31
__device__ __forceinline__ v16bf frag_b(const bf16* Bs, int lane) {
    const bf16* p = Bs + (lane & 15) * LDT + ((lane >> 4) << 3);
    v16bf b;
    ((uint4*)&b)[0] = *(const uint4*)p;
    ((uint4*)&b)[1] = *(const uint4*)(p + 16);
    return b;
}

// Stage a 64-row x 32-col bf16 tile (K-major global) into LDS. 128 threads, 32B each.
// Async path: 2 GLOBAL_LOAD_ASYNC_TO_LDS_B128 per tile (ASYNCcnt += 2 per wave).
__device__ __forceinline__ void stage_tile(const bf16* __restrict__ g, int ld,
                                           int rowbase, int k0, bf16* S, int tid) {
    int row = tid >> 1;
    int seg = (tid & 1) << 4;
    const bf16* src = g + (size_t)(rowbase + row) * ld + k0 + seg;
    bf16* dst = S + row * LDT + seg;
#if USE_ASYNC
    __builtin_amdgcn_global_load_async_to_lds_b128((AS1 v4i*)src,       (AS3 v4i*)dst,       0, 0);
    __builtin_amdgcn_global_load_async_to_lds_b128((AS1 v4i*)(src + 8), (AS3 v4i*)(dst + 8), 0, 0);
#else
    *(uint4*)dst       = *(const uint4*)src;
    *(uint4*)(dst + 8) = *(const uint4*)(src + 8);
#endif
}

// ---------------- fused gate+up SwiGLU GEMM (bf16 WMMA, f32 acc) ----------------
// Block tile 64x64, 4 waves; each wave owns a 32x32 quadrant:
// 2 A-frags + 2 Bg-frags + 2 Bu-frags -> 8 WMMAs per K-step (best LDS-read/WMMA ratio).
__global__ __launch_bounds__(128)
void gateup_kernel(const bf16* __restrict__ Aall, const bf16* __restrict__ BgAll,
                   const bf16* __restrict__ BuAll, bf16* __restrict__ OutAll,
                   const int* __restrict__ counts, int K, int ldo,
                   long aStride, long bStride, long oStride, int useCounts)
{
    __shared__ alignas(16) bf16 smA [2][64 * LDT];
    __shared__ alignas(16) bf16 smBg[2][64 * LDT];
    __shared__ alignas(16) bf16 smBu[2][64 * LDT];

    int e     = blockIdx.z;
    int mbase = blockIdx.y * 64;
    int nbase = blockIdx.x * 64;
    if (useCounts) {
        int cnt = counts[e]; if (cnt > CAP_E) cnt = CAP_E;
        if (mbase >= cnt) return;           // uniform early-exit, EXEC stays all-ones
    }
    const bf16* A  = Aall  + (size_t)e * aStride;
    const bf16* Bg = BgAll + (size_t)e * bStride;
    const bf16* Bu = BuAll + (size_t)e * bStride;
    bf16*       Out = OutAll + (size_t)e * oStride;

    int tid  = threadIdx.x;
    int lane = tid & 31;
    int wv   = tid >> 5;
    int wm   = (wv & 1) << 5;          // wave's row quadrant (0/32)
    int wn   = (wv >> 1) << 5;         // wave's col quadrant (0/32)

    v8f accg[2][2], accu[2][2];
#pragma unroll
    for (int i = 0; i < 2; ++i)
#pragma unroll
        for (int j = 0; j < 2; ++j) { accg[i][j] = (v8f)(0.0f); accu[i][j] = (v8f)(0.0f); }

    int nk = K >> 5;
    stage_tile(A,  K, mbase, 0, smA [0], tid);
    stage_tile(Bg, K, nbase, 0, smBg[0], tid);
    stage_tile(Bu, K, nbase, 0, smBu[0], tid);

    for (int ki = 0; ki < nk; ++ki) {
        int cur = ki & 1;
        if (ki + 1 < nk) {
            int nxt = cur ^ 1, k0 = (ki + 1) << 5;
            stage_tile(A,  K, mbase, k0, smA [nxt], tid);
            stage_tile(Bg, K, nbase, k0, smBg[nxt], tid);
            stage_tile(Bu, K, nbase, k0, smBu[nxt], tid);
#if USE_ASYNC
            __builtin_amdgcn_s_wait_asynccnt(6);   // tile ki's 6 loads done; k+1's may fly
#endif
        }
#if USE_ASYNC
        else {
            __builtin_amdgcn_s_wait_asynccnt(0);
        }
#endif
        __syncthreads();

        v16bf a0  = frag_a(smA [cur] + (wm     ) * LDT, lane);
        v16bf a1  = frag_a(smA [cur] + (wm + 16) * LDT, lane);
        v16bf bg0 = frag_b(smBg[cur] + (wn     ) * LDT, lane);
        v16bf bg1 = frag_b(smBg[cur] + (wn + 16) * LDT, lane);
        v16bf bu0 = frag_b(smBu[cur] + (wn     ) * LDT, lane);
        v16bf bu1 = frag_b(smBu[cur] + (wn + 16) * LDT, lane);

        accg[0][0] = __builtin_amdgcn_wmma_f32_16x16x32_bf16(false, a0, false, bg0, (short)0, accg[0][0], false, false);
        accg[0][1] = __builtin_amdgcn_wmma_f32_16x16x32_bf16(false, a0, false, bg1, (short)0, accg[0][1], false, false);
        accg[1][0] = __builtin_amdgcn_wmma_f32_16x16x32_bf16(false, a1, false, bg0, (short)0, accg[1][0], false, false);
        accg[1][1] = __builtin_amdgcn_wmma_f32_16x16x32_bf16(false, a1, false, bg1, (short)0, accg[1][1], false, false);
        accu[0][0] = __builtin_amdgcn_wmma_f32_16x16x32_bf16(false, a0, false, bu0, (short)0, accu[0][0], false, false);
        accu[0][1] = __builtin_amdgcn_wmma_f32_16x16x32_bf16(false, a0, false, bu1, (short)0, accu[0][1], false, false);
        accu[1][0] = __builtin_amdgcn_wmma_f32_16x16x32_bf16(false, a1, false, bu0, (short)0, accu[1][0], false, false);
        accu[1][1] = __builtin_amdgcn_wmma_f32_16x16x32_bf16(false, a1, false, bu1, (short)0, accu[1][1], false, false);

        __syncthreads();
    }

    // epilogue: h = silu(g) * u, store bf16
    int row0 = mbase + wm + ((lane >> 4) << 3);
    int col0 = nbase + wn + (lane & 15);
#pragma unroll
    for (int ai = 0; ai < 2; ++ai) {
#pragma unroll
        for (int bi = 0; bi < 2; ++bi) {
#pragma unroll
            for (int r = 0; r < 8; ++r) {
                float g  = accg[ai][bi][r];
                float u  = accu[ai][bi][r];
                float hv = (g / (1.0f + __expf(-g))) * u;
                Out[(size_t)(row0 + ai * 16 + r) * ldo + col0 + bi * 16] = (bf16)hv;
            }
        }
    }
}

// ---------------- down-projection GEMM (+ scatter/combine for routed path) ----------------
__global__ __launch_bounds__(128)
void down_kernel(const bf16* __restrict__ Aall, const bf16* __restrict__ Ball,
                 float* __restrict__ Y, const int* __restrict__ counts,
                 const int* __restrict__ tokmap, const float* __restrict__ tokw,
                 int K, long aStride, long bStride, int routed)
{
    __shared__ alignas(16) bf16 smA[2][64 * LDT];
    __shared__ alignas(16) bf16 smB[2][64 * LDT];

    int e     = blockIdx.z;
    int mbase = blockIdx.y * 64;
    int nbase = blockIdx.x * 64;
    int cnt   = 1 << 30;
    if (routed) {
        cnt = counts[e]; if (cnt > CAP_E) cnt = CAP_E;
        if (mbase >= cnt) return;
    }
    const bf16* A  = Aall + (size_t)e * aStride;
    const bf16* Bt = Ball + (size_t)e * bStride;

    int tid  = threadIdx.x;
    int lane = tid & 31;
    int wv   = tid >> 5;
    int wm   = (wv & 1) << 5;
    int wn   = (wv >> 1) << 5;

    v8f acc[2][2];
#pragma unroll
    for (int i = 0; i < 2; ++i)
#pragma unroll
        for (int j = 0; j < 2; ++j) acc[i][j] = (v8f)(0.0f);

    int nk = K >> 5;
    stage_tile(A,  K, mbase, 0, smA[0], tid);
    stage_tile(Bt, K, nbase, 0, smB[0], tid);

    for (int ki = 0; ki < nk; ++ki) {
        int cur = ki & 1;
        if (ki + 1 < nk) {
            int nxt = cur ^ 1, k0 = (ki + 1) << 5;
            stage_tile(A,  K, mbase, k0, smA[nxt], tid);
            stage_tile(Bt, K, nbase, k0, smB[nxt], tid);
#if USE_ASYNC
            __builtin_amdgcn_s_wait_asynccnt(4);
#endif
        }
#if USE_ASYNC
        else {
            __builtin_amdgcn_s_wait_asynccnt(0);
        }
#endif
        __syncthreads();

        v16bf a0 = frag_a(smA[cur] + (wm     ) * LDT, lane);
        v16bf a1 = frag_a(smA[cur] + (wm + 16) * LDT, lane);
        v16bf b0 = frag_b(smB[cur] + (wn     ) * LDT, lane);
        v16bf b1 = frag_b(smB[cur] + (wn + 16) * LDT, lane);

        acc[0][0] = __builtin_amdgcn_wmma_f32_16x16x32_bf16(false, a0, false, b0, (short)0, acc[0][0], false, false);
        acc[0][1] = __builtin_amdgcn_wmma_f32_16x16x32_bf16(false, a0, false, b1, (short)0, acc[0][1], false, false);
        acc[1][0] = __builtin_amdgcn_wmma_f32_16x16x32_bf16(false, a1, false, b0, (short)0, acc[1][0], false, false);
        acc[1][1] = __builtin_amdgcn_wmma_f32_16x16x32_bf16(false, a1, false, b1, (short)0, acc[1][1], false, false);

        __syncthreads();
    }

    int row0 = mbase + wm + ((lane >> 4) << 3);
    int col0 = nbase + wn + (lane & 15);
#pragma unroll
    for (int ai = 0; ai < 2; ++ai) {
#pragma unroll
        for (int bi = 0; bi < 2; ++bi) {
#pragma unroll
            for (int r = 0; r < 8; ++r) {
                int   gm = row0 + ai * 16 + r;
                int   c  = col0 + bi * 16;
                float v  = acc[ai][bi][r];
                if (routed) {
                    if (gm < cnt) {
                        int   slot = e * CAP_E + gm;
                        int   tok  = tokmap[slot];
                        float w    = tokw[slot];
                        __hip_atomic_fetch_add(&Y[(size_t)tok * H_DIM + c], v * w,
                                               __ATOMIC_RELAXED, __HIP_MEMORY_SCOPE_AGENT);
                    }
                } else {
                    Y[(size_t)gm * H_DIM + c] = v;   // shared expert writes y first
                }
            }
        }
    }
}

// ---------------- router: logits -> softmax -> top-6 ----------------
__global__ __launch_bounds__(64)
void gate_topk_kernel(const float* __restrict__ x, const float* __restrict__ gw,
                      int* __restrict__ tidx, float* __restrict__ tw)
{
    __shared__ alignas(16) float xs[H_DIM];
    __shared__ float ls[NE];
    int t = blockIdx.x, tid = threadIdx.x;
    for (int i = tid; i < H_DIM; i += 64) xs[i] = x[(size_t)t * H_DIM + i];
    __syncthreads();
    float acc = 0.f;
    const float* wr = gw + (size_t)tid * H_DIM;
    for (int h = 0; h < H_DIM; h += 4) {
        float4 xv = *(const float4*)(xs + h);
        float4 wv = *(const float4*)(wr + h);
        acc += xv.x * wv.x + xv.y * wv.y + xv.z * wv.z + xv.w * wv.w;
    }
    ls[tid] = acc;
    __syncthreads();
    if (tid == 0) {
        float m = ls[0];
        for (int i = 1; i < NE; ++i) m = fmaxf(m, ls[i]);
        float s = 0.f;
        for (int i = 0; i < NE; ++i) { ls[i] = __expf(ls[i] - m); s += ls[i]; }
        float inv = 1.f / s;
        for (int k = 0; k < TOPK; ++k) {
            int bi = 0; float bv = -1.f;
            for (int i = 0; i < NE; ++i) if (ls[i] > bv) { bv = ls[i]; bi = i; }
            tidx[t * TOPK + k] = bi;
            tw  [t * TOPK + k] = bv * inv;      // SCALE == 1.0
            ls[bi] = -2.f;
        }
    }
}

// ---------------- dispatch: atomic slot assignment + bf16 token copy ----------------
__global__ __launch_bounds__(256)
void dispatch_kernel(const bf16* __restrict__ xb, const int* __restrict__ tidx,
                     const float* __restrict__ tw, int* __restrict__ counts,
                     int* __restrict__ tokmap, float* __restrict__ tokw,
                     bf16* __restrict__ bufA)
{
    int id = blockIdx.x;
    int t  = id / TOPK, k = id - t * TOPK;
    __shared__ int sp, se;
    if (threadIdx.x == 0) {
        int e = tidx[t * TOPK + k];
        int p = atomicAdd(&counts[e], 1);
        se = e; sp = p;
        if (p < CAP_E) {
            tokmap[e * CAP_E + p] = t;
            tokw  [e * CAP_E + p] = tw[t * TOPK + k];
        }
    }
    __syncthreads();
    int p = sp, e = se;
    if (p >= CAP_E) return;                      // capacity overflow: drop (matches ref stats)
    const uint4* src = (const uint4*)(xb + (size_t)t * H_DIM);
    uint4*       dst = (uint4*)(bufA + ((size_t)e * CAP_E + p) * H_DIM);
    dst[threadIdx.x] = src[threadIdx.x];         // 256 lanes x 16B = 4KB row
}

// ---------------- f32 -> bf16 convert (layout preserved) ----------------
__global__ __launch_bounds__(256)
void cvt_f32_bf16(const float* __restrict__ in, bf16* __restrict__ out, long n)
{
    long base = ((long)blockIdx.x * 256 + threadIdx.x) * 4;
    if (base + 3 < n) {
        float4 v = *(const float4*)(in + base);
        union { bf16 h[4]; uint2 u; } t;
        t.h[0] = (bf16)v.x; t.h[1] = (bf16)v.y; t.h[2] = (bf16)v.z; t.h[3] = (bf16)v.w;
        *(uint2*)(out + base) = t.u;
    }
}

// ---------------- batched tiled transpose-convert: f32 [z][R][C] -> bf16 [z][C][R] ----------------
__global__ __launch_bounds__(256)
void transpose_cvt(const float* __restrict__ in, bf16* __restrict__ out, int R, int C)
{
    __shared__ float tile[32][33];
    int z  = blockIdx.z;
    int r0 = blockIdx.y * 32, c0 = blockIdx.x * 32;
    const float* I = in  + (size_t)z * R * C;
    bf16*        O = out + (size_t)z * R * C;
    int c = threadIdx.x & 31, rr = threadIdx.x >> 5;
#pragma unroll
    for (int i = 0; i < 4; ++i) {
        int r = rr + i * 8;
        tile[r][c] = I[(size_t)(r0 + r) * C + c0 + c];
    }
    __syncthreads();
#pragma unroll
    for (int i = 0; i < 4; ++i) {
        int r = rr + i * 8;
        O[(size_t)(c0 + r) * R + r0 + c] = (bf16)tile[c][r];
    }
}

__global__ void zero_counts_kernel(int* counts) {
    if (threadIdx.x < NE) counts[threadIdx.x] = 0;
}

// ---------------- launcher ----------------
extern "C" void kernel_launch(void* const* d_in, const int* in_sizes, int n_in,
                              void* d_out, int out_size, void* d_ws, size_t ws_size,
                              hipStream_t stream)
{
    const float* x       = (const float*)d_in[0];
    const float* gate_w  = (const float*)d_in[1];
    const float* w_gate  = (const float*)d_in[2];
    const float* w_up    = (const float*)d_in[3];
    const float* w_down  = (const float*)d_in[4];
    const float* sh_gate = (const float*)d_in[5];
    const float* sh_up   = (const float*)d_in[6];
    const float* sh_down = (const float*)d_in[7];
    float* y = (float*)d_out;

    // workspace carve (256B aligned)
    char* p = (char*)d_ws; size_t off = 0;
    auto carve = [&](size_t bytes) -> char* {
        char* r = p + off; off += (bytes + 255) & ~(size_t)255; return r;
    };
    bf16*  xb    = (bf16*) carve((size_t)NTOK * H_DIM * 2);
    bf16*  wgT   = (bf16*) carve((size_t)NE * F_DIM * H_DIM * 2);   // [E][F][H]
    bf16*  wuT   = (bf16*) carve((size_t)NE * F_DIM * H_DIM * 2);   // [E][F][H]
    bf16*  wdT   = (bf16*) carve((size_t)NE * H_DIM * F_DIM * 2);   // [E][H][F]
    bf16*  shg   = (bf16*) carve((size_t)FSH * H_DIM * 2);          // [FSH][H] (already K-major)
    bf16*  shu   = (bf16*) carve((size_t)FSH * H_DIM * 2);
    bf16*  shd   = (bf16*) carve((size_t)H_DIM * FSH * 2);          // [H][FSH] (already K-major)
    int*   tidx  = (int*)  carve((size_t)NTOK * TOPK * 4);
    float* tw    = (float*)carve((size_t)NTOK * TOPK * 4);
    int*   cnts  = (int*)  carve((size_t)NE * 4);
    int*   tokmap= (int*)  carve((size_t)NE * CAP_E * 4);
    float* tokw  = (float*)carve((size_t)NE * CAP_E * 4);
    bf16*  bufA  = (bf16*) carve((size_t)NE * CAP_E * H_DIM * 2);
    bf16*  Hh    = (bf16*) carve((size_t)NE * CAP_E * F_DIM * 2);
    bf16*  hs    = (bf16*) carve((size_t)NTOK * FSH * 2);
    (void)ws_size; (void)in_sizes; (void)n_in; (void)out_size;

    // 1) precision/layout conversion passes
    {
        long n;
        n = (long)NTOK * H_DIM;
        cvt_f32_bf16<<<dim3((unsigned)((n / 4 + 255) / 256)), 256, 0, stream>>>(x, xb, n);
        n = (long)FSH * H_DIM;
        cvt_f32_bf16<<<dim3((unsigned)((n / 4 + 255) / 256)), 256, 0, stream>>>(sh_gate, shg, n);
        cvt_f32_bf16<<<dim3((unsigned)((n / 4 + 255) / 256)), 256, 0, stream>>>(sh_up, shu, n);
        n = (long)H_DIM * FSH;
        cvt_f32_bf16<<<dim3((unsigned)((n / 4 + 255) / 256)), 256, 0, stream>>>(sh_down, shd, n);
    }
    transpose_cvt<<<dim3(F_DIM / 32, H_DIM / 32, NE), 256, 0, stream>>>(w_gate, wgT, H_DIM, F_DIM);
    transpose_cvt<<<dim3(F_DIM / 32, H_DIM / 32, NE), 256, 0, stream>>>(w_up,   wuT, H_DIM, F_DIM);
    transpose_cvt<<<dim3(H_DIM / 32, F_DIM / 32, NE), 256, 0, stream>>>(w_down, wdT, F_DIM, H_DIM);

    // 2) router + dispatch
    gate_topk_kernel<<<NTOK, 64, 0, stream>>>(x, gate_w, tidx, tw);
    zero_counts_kernel<<<1, 64, 0, stream>>>(cnts);
    dispatch_kernel<<<NTOK * TOPK, 256, 0, stream>>>(xb, tidx, tw, cnts, tokmap, tokw, bufA);

    // 3) shared expert (writes y) — must precede routed down (atomic adds)
    gateup_kernel<<<dim3(FSH / 64, NTOK / 64, 1), 128, 0, stream>>>(
        xb, shg, shu, hs, nullptr, H_DIM, FSH, 0, 0, 0, 0);
    down_kernel<<<dim3(H_DIM / 64, NTOK / 64, 1), 128, 0, stream>>>(
        hs, shd, y, nullptr, nullptr, nullptr, FSH, 0, 0, 0);

    // 4) routed experts (grouped GEMM, early-exit past counts[e])
    gateup_kernel<<<dim3(F_DIM / 64, CAP_E / 64, NE), 128, 0, stream>>>(
        bufA, wgT, wuT, Hh, cnts, H_DIM, F_DIM,
        (long)CAP_E * H_DIM, (long)F_DIM * H_DIM, (long)CAP_E * F_DIM, 1);
    down_kernel<<<dim3(H_DIM / 64, CAP_E / 64, NE), 128, 0, stream>>>(
        Hh, wdT, y, cnts, tokmap, tokw, F_DIM,
        (long)CAP_E * F_DIM, (long)H_DIM * F_DIM, 1);
}
